// HIDE_38319698215693
// MI455X (gfx1250) — matrix operand order repaired and checked by reading, not compiled, not run on hardware
//
#include <hip/hip_runtime.h>

// ---------------------------------------------------------------------------
// Problem constants (from the reference)
// ---------------------------------------------------------------------------
#define NUM_NODE 50000
#define DIM      128
#define IMG_DIM  1000
#define TXT_DIM  768
#define BB       512
#define LL       50
#define NTILES   8                 // DIM / 16
#define IMG_KSTEPS 32              // ceil(1000/32) ; last step is the tail
#define TXT_KSTEPS 24              // 768/32 exact
#define MTILES   (NUM_NODE / 16)   // 3125

typedef __attribute__((ext_vector_type(16))) __bf16          v16bf;
typedef __attribute__((ext_vector_type(16))) unsigned short  v16u;
typedef __attribute__((ext_vector_type(8)))  float           v8f;

// fp32 -> bf16, round-to-nearest-even (bit manipulation, no __bf16 arithmetic)
__device__ __forceinline__ unsigned short f2bf(float f) {
  unsigned int u = __float_as_uint(f);
  u += 0x7FFFu + ((u >> 16) & 1u);
  return (unsigned short)(u >> 16);
}

// ---------------------------------------------------------------------------
// Pack a (K x 128) fp32 weight matrix into the CDNA5 WMMA B-operand layout:
//   B tile is 32(K) x 16(N); lane = half*16 + n; element j<8 holds K = half*8+j, j>=8 holds K = 16+half*8+(j-8).
// packed layout: [kstep][ntile][lane][16 bf16]  (32B per lane, coalesced)
// ---------------------------------------------------------------------------
__global__ void pack_w_kernel(const float* __restrict__ W, int K,
                              unsigned short* __restrict__ packed) {
  int tile  = blockIdx.x;            // kstep * NTILES + ntile
  int kstep = tile >> 3;
  int nt    = tile & 7;
  int lane  = threadIdx.x;           // 0..31
  int hf    = lane >> 4;
  int n     = (lane & 15) + nt * 16;
  unsigned short* outp = packed + ((size_t)tile * 32 + lane) * 16;
  int k0 = kstep * 32 + hf * 8;
#pragma unroll
  for (int j = 0; j < 16; ++j) {
    int k = k0 + ((j < 8) ? j : 16 + (j - 8));
    float f = (k < K) ? W[(size_t)k * DIM + n] : 0.f;
    outp[j] = f2bf(f);
  }
}

// Zero the pad row (row 0) of the three padded tables.
__global__ void zero_pad_kernel(float* imgE, float* txtE, float* itemE) {
  int d = threadIdx.x;
  imgE[d] = 0.f; txtE[d] = 0.f; itemE[d] = 0.f;
}

// ---------------------------------------------------------------------------
// A-operand loaders: 16x32 fp32 tile -> bf16 WMMA A layout.
//   lane = half*16 + m;  elements 0..7:  K = k0 + half*8 + j
//                        elements 8..15: K = k0 + 16 + half*8 + j
// Full variant: unguarded float4 loads (rows are 16B aligned: 1000*4, 768*4).
// Tail variant: address-clamped loads + value select (no exec branching).
// ---------------------------------------------------------------------------
__device__ __forceinline__ v16bf load_a_bf16_full(const float* __restrict__ rp,
                                                  int k0, int hf) {
  const float4* p1 = (const float4*)(rp + k0 + hf * 8);        // K: kb1..kb1+7
  const float4* p2 = (const float4*)(rp + k0 + 16 + hf * 8);   // K: kb2..kb2+7
  __builtin_prefetch(rp + k0 + 64, 0, 3);   // stream-ahead, WGP scope
  float4 a0 = p1[0], a1 = p1[1];
  float4 b0 = p2[0], b1 = p2[1];
  float t[16] = {a0.x, a0.y, a0.z, a0.w, a1.x, a1.y, a1.z, a1.w,
                 b0.x, b0.y, b0.z, b0.w, b1.x, b1.y, b1.z, b1.w};
  v16u u;
#pragma unroll
  for (int j = 0; j < 16; ++j) u[j] = f2bf(t[j]);
  return __builtin_bit_cast(v16bf, u);
}

__device__ __forceinline__ v16bf load_a_bf16_tail(const float* __restrict__ rp,
                                                  int K, int k0, int hf) {
  int kb1 = k0 + hf * 8;
  int kb2 = k0 + 16 + hf * 8;
  v16u u;
#pragma unroll
  for (int j = 0; j < 8; ++j) {
    int k1 = kb1 + j, k2 = kb2 + j;
    float f1 = rp[(k1 < K) ? k1 : 0];     // clamped addr -> unconditional load
    float f2 = rp[(k2 < K) ? k2 : 0];
    u[j]     = (k1 < K) ? f2bf(f1) : (unsigned short)0;
    u[8 + j] = (k2 < K) ? f2bf(f2) : (unsigned short)0;
  }
  return __builtin_bit_cast(v16bf, u);
}

// ---------------------------------------------------------------------------
// Fused table projection (the HBM-streaming hot loop):
//   img_all = img_table @ img_W + img_b        (bf16 WMMA, fp32 acc)
//   txt_all = txt_table @ txt_W + txt_b
//   itemE   = emb + 0.1*img_all + 0.15*txt_all  (all written with +1 row pad)
// One wave owns a 16-row x 128-col output tile (8 v8f accumulators).
// ---------------------------------------------------------------------------
__global__ __launch_bounds__(256) void table_proj_kernel(
    const float* __restrict__ img_table, const float* __restrict__ txt_table,
    const float* __restrict__ emb_table,
    const unsigned short* __restrict__ pImgW,
    const unsigned short* __restrict__ pTxtW,
    const float* __restrict__ img_b, const float* __restrict__ txt_b,
    float* __restrict__ imgE, float* __restrict__ txtE,
    float* __restrict__ itemE) {
  int wave  = threadIdx.x >> 5;
  int lane  = threadIdx.x & 31;
  int mtile = blockIdx.x * 8 + wave;
  if (mtile >= MTILES) return;               // wave-uniform exit (no barriers)
  int row0 = mtile * 16;
  int hf   = lane >> 4;
  int mr   = lane & 15;

  v8f zero = {0.f,0.f,0.f,0.f,0.f,0.f,0.f,0.f};
  v8f acc[NTILES], accItem[NTILES];
#pragma unroll
  for (int nt = 0; nt < NTILES; ++nt) acc[nt] = zero;

  // ---- IMG GEMM: K=1000 -> 31 full 32-K steps + 1 tail step ----
  {
    const float* rp = img_table + (size_t)(row0 + mr) * IMG_DIM;
    for (int ks = 0; ks < IMG_KSTEPS - 1; ++ks) {
      v16bf a = load_a_bf16_full(rp, ks * 32, hf);
#pragma unroll
      for (int nt = 0; nt < NTILES; ++nt) {
        v16u bu = *(const v16u*)(pImgW + ((size_t)(ks * NTILES + nt) * 32 + lane) * 16);
        v16bf b = __builtin_bit_cast(v16bf, bu);
        acc[nt] = __builtin_amdgcn_wmma_f32_16x16x32_bf16(
            false, a, false, b, (short)0, acc[nt], false, false);
      }
    }
    {
      const int ks = IMG_KSTEPS - 1;
      v16bf a = load_a_bf16_tail(rp, IMG_DIM, ks * 32, hf);
#pragma unroll
      for (int nt = 0; nt < NTILES; ++nt) {
        v16u bu = *(const v16u*)(pImgW + ((size_t)(ks * NTILES + nt) * 32 + lane) * 16);
        v16bf b = __builtin_bit_cast(v16bf, bu);
        acc[nt] = __builtin_amdgcn_wmma_f32_16x16x32_bf16(
            false, a, false, b, (short)0, acc[nt], false, false);
      }
    }
  }
  // img epilogue: bias, store imgE (+1 pad row), seed item accumulator
#pragma unroll
  for (int nt = 0; nt < NTILES; ++nt) {
#pragma unroll
    for (int r = 0; r < 8; ++r) {
      int row = row0 + r + hf * 8;
      int col = nt * 16 + mr;
      float v = acc[nt][r] + img_b[col];
      imgE[(size_t)(row + 1) * DIM + col] = v;
      accItem[nt][r] = 0.1f * v;
    }
    acc[nt] = zero;
  }

  // ---- TXT GEMM: K=768 = 24 full steps, no tail ----
  {
    const float* rp = txt_table + (size_t)(row0 + mr) * TXT_DIM;
    for (int ks = 0; ks < TXT_KSTEPS; ++ks) {
      v16bf a = load_a_bf16_full(rp, ks * 32, hf);
#pragma unroll
      for (int nt = 0; nt < NTILES; ++nt) {
        v16u bu = *(const v16u*)(pTxtW + ((size_t)(ks * NTILES + nt) * 32 + lane) * 16);
        v16bf b = __builtin_bit_cast(v16bf, bu);
        acc[nt] = __builtin_amdgcn_wmma_f32_16x16x32_bf16(
            false, a, false, b, (short)0, acc[nt], false, false);
      }
    }
  }
  // txt epilogue + item fuse
#pragma unroll
  for (int nt = 0; nt < NTILES; ++nt) {
#pragma unroll
    for (int r = 0; r < 8; ++r) {
      int row = row0 + r + hf * 8;
      int col = nt * 16 + mr;
      float v = acc[nt][r] + txt_b[col];
      txtE[(size_t)(row + 1) * DIM + col] = v;
      float it = emb_table[(size_t)row * DIM + col] + accItem[nt][r] + 0.15f * v;
      itemE[(size_t)(row + 1) * DIM + col] = it;
    }
  }
}

// ---------------------------------------------------------------------------
// Masked session means over L items (gathers from padded tables).
// ---------------------------------------------------------------------------
__global__ void session_mean_kernel(const int* __restrict__ item,
                                    const int* __restrict__ mask,
                                    const float* __restrict__ imgE,
                                    const float* __restrict__ txtE,
                                    const float* __restrict__ itemE,
                                    float* sImg, float* sTxt, float* sItem) {
  int b = blockIdx.x, d = threadIdx.x;
  float si = 0.f, st = 0.f, se = 0.f, dn = 0.f;
  for (int l = 0; l < LL; ++l) {
    int idx = item[b * LL + l];
    float m = (float)mask[b * LL + l];
    dn += m;
    si += imgE [(size_t)idx * DIM + d] * m;
    st += txtE [(size_t)idx * DIM + d] * m;
    se += itemE[(size_t)idx * DIM + d] * m;
  }
  sImg [b * DIM + d] = si / dn;
  sTxt [b * DIM + d] = st / dn;
  sItem[b * DIM + d] = se / dn;
}

__device__ __forceinline__ float block_reduce_sum128(float v, float* red) {
  int d = threadIdx.x;
  red[d] = v;
  __syncthreads();
  for (int s = 64; s > 0; s >>= 1) {
    if (d < s) red[d] += red[d + s];
    __syncthreads();
  }
  float r = red[0];
  __syncthreads();
  return r;
}

// ---------------------------------------------------------------------------
// Gate / qc / softmax fusion -> session_diff.  One block (128 thr) per row.
// ---------------------------------------------------------------------------
__global__ void fusion_kernel(const float* __restrict__ sImg,
                              const float* __restrict__ sTxt,
                              const float* __restrict__ sItem,
                              const float* __restrict__ gvW, const float* __restrict__ gvb,
                              const float* __restrict__ gtW, const float* __restrict__ gtb,
                              const float* __restrict__ qcW1, const float* __restrict__ qcb1,
                              const float* __restrict__ qcW2,
                              float* __restrict__ sDiff) {
  int b = blockIdx.x, d = threadIdx.x;
  __shared__ float xi[DIM], xt[DIM], xe[DIM], sid[DIM], stdv[DIM], red[DIM];
  xi[d] = sImg[b * DIM + d];
  xt[d] = sTxt[b * DIM + d];
  xe[d] = sItem[b * DIM + d];
  __syncthreads();

  // gate_v(2*img), gate_t(2*txt)
  float a1 = gvb[d], a2 = gtb[d];
  for (int k = 0; k < DIM; ++k) {
    a1 += 2.f * xi[k] * gvW[k * DIM + d];
    a2 += 2.f * xt[k] * gtW[k * DIM + d];
  }
  float gv = 1.f / (1.f + expf(-a1));
  float gt = 1.f / (1.f + expf(-a2));
  sid[d]  = xe[d] * gv;
  stdv[d] = xe[d] * gt;
  __syncthreads();

  // qc(sid), qc(std): tanh(x@W1+b1) @ W2
  float q1 = qcb1[d], q2 = qcb1[d];
  for (int k = 0; k < DIM; ++k) {
    float w = qcW1[k * DIM + d];
    q1 += sid[k] * w;
    q2 += stdv[k] * w;
  }
  float w2d  = qcW2[d];
  float qsid = block_reduce_sum128(tanhf(q1) * w2d, red);
  float qstd = block_reduce_sum128(tanhf(q2) * w2d, red);
  float mx = fmaxf(qsid, qstd);
  float e0 = expf(qsid - mx), e1 = expf(qstd - mx);
  float inv = 1.f / (e0 + e1);
  float common = (e0 * inv) * sid[d] + (e1 * inv) * stdv[d];

  // gate_v(item), gate_t(item)
  float b1v = gvb[d], b2v = gtb[d];
  for (int k = 0; k < DIM; ++k) {
    float x = xe[k];
    b1v += x * gvW[k * DIM + d];
    b2v += x * gtW[k * DIM + d];
  }
  float gvi = 1.f / (1.f + expf(-b1v));
  float gti = 1.f / (1.f + expf(-b2v));
  float sep_i = gvi * (sid[d]  - common);
  float sep_t = gti * (stdv[d] - common);
  float fus = (sep_i + sep_t + common) * (1.f / 3.f);
  sDiff[b * DIM + d] = xe[d] + xi[d] + xt[d] + fus;
}

// h0 = itemE[inputs]  (written straight into d_out)
__global__ void gather_h_kernel(const int* __restrict__ inputs,
                                const float* __restrict__ itemE,
                                float* __restrict__ h) {
  int t = blockIdx.x;            // b*L + l
  int d = threadIdx.x;
  int idx = inputs[t];
  h[(size_t)t * DIM + d] = itemE[(size_t)idx * DIM + d];
}

// ---------------------------------------------------------------------------
// 2-layer hypergraph propagation, fully in LDS (62 KB of the 320 KB WGP LDS),
// one block per batch element, in-place on d_out.
// ---------------------------------------------------------------------------
__global__ __launch_bounds__(256) void propagate_kernel(
    const float* __restrict__ Hs, const float* __restrict__ sDiff,
    float* __restrict__ h) {
  int b = blockIdx.x, tid = threadIdx.x;
  __shared__ float hs_s[LL * LL];        // 10000 B
  __shared__ float h_s[LL * DIM];        // 25600 B
  __shared__ float e_s[LL * DIM];        // 25600 B
  __shared__ float s_s[DIM];
  __shared__ float dege[LL], degn[LL];

  for (int i = tid; i < LL * LL; i += 256) hs_s[i] = Hs[(size_t)b * LL * LL + i];
  for (int i = tid; i < LL * DIM; i += 256) h_s[i] = h[(size_t)b * LL * DIM + i];
  if (tid < DIM) s_s[tid] = sDiff[b * DIM + tid];
  __syncthreads();
  if (tid < LL) {
    float de = 0.f, dn = 0.f;
    for (int l = 0; l < LL; ++l) de += hs_s[l * LL + tid];   // sum over axis 1
    for (int e = 0; e < LL; ++e) dn += hs_s[tid * LL + e];   // sum over axis 2
    dege[tid] = de + 1e-8f;
    degn[tid] = dn + 1e-8f;
  }
  __syncthreads();

  for (int layer = 0; layer < 2; ++layer) {
    for (int idx = tid; idx < LL * DIM; idx += 256) {
      int ee = idx >> 7, d = idx & 127;
      float sum = 0.f;
      for (int l = 0; l < LL; ++l) sum += hs_s[l * LL + ee] * h_s[l * DIM + d];
      e_s[idx] = sum / dege[ee];
    }
    __syncthreads();
    for (int idx = tid; idx < LL * DIM; idx += 256) {
      int l = idx >> 7, d = idx & 127;
      float sum = 0.f;
      for (int e2 = 0; e2 < LL; ++e2) sum += hs_s[l * LL + e2] * e_s[e2 * DIM + d];
      h_s[idx] = sum / degn[l] + s_s[d];
    }
    __syncthreads();
  }
  for (int i = tid; i < LL * DIM; i += 256) h[(size_t)b * LL * DIM + i] = h_s[i];
}

// ---------------------------------------------------------------------------
extern "C" void kernel_launch(void* const* d_in, const int* in_sizes, int n_in,
                              void* d_out, int out_size, void* d_ws, size_t ws_size,
                              hipStream_t stream) {
  const int*   inputs  = (const int*)  d_in[0];
  const int*   item    = (const int*)  d_in[1];
  const int*   mask    = (const int*)  d_in[2];
  const float* Hs      = (const float*)d_in[3];
  const float* emb     = (const float*)d_in[4];
  const float* img_t   = (const float*)d_in[5];
  const float* txt_t   = (const float*)d_in[6];
  const float* img_W   = (const float*)d_in[7];
  const float* img_b   = (const float*)d_in[8];
  const float* txt_W   = (const float*)d_in[9];
  const float* txt_b   = (const float*)d_in[10];
  const float* gvW     = (const float*)d_in[11];
  const float* gvb     = (const float*)d_in[12];
  const float* gtW     = (const float*)d_in[13];
  const float* gtb     = (const float*)d_in[14];
  const float* qcW1    = (const float*)d_in[15];
  const float* qcb1    = (const float*)d_in[16];
  const float* qcW2    = (const float*)d_in[17];
  float* out = (float*)d_out;

  char* ws = (char*)d_ws;
  size_t off = 0;
  auto alloc = [&](size_t bytes) -> char* {
    char* p = ws + off;
    off += (bytes + 255) & ~(size_t)255;
    return p;
  };
  float* imgE  = (float*)alloc((size_t)(NUM_NODE + 1) * DIM * 4);
  float* txtE  = (float*)alloc((size_t)(NUM_NODE + 1) * DIM * 4);
  float* itemE = (float*)alloc((size_t)(NUM_NODE + 1) * DIM * 4);
  float* sImg  = (float*)alloc((size_t)BB * DIM * 4);
  float* sTxt  = (float*)alloc((size_t)BB * DIM * 4);
  float* sItem = (float*)alloc((size_t)BB * DIM * 4);
  float* sDiff = (float*)alloc((size_t)BB * DIM * 4);
  unsigned short* pImgW = (unsigned short*)alloc((size_t)IMG_KSTEPS * NTILES * 32 * 16 * 2);
  unsigned short* pTxtW = (unsigned short*)alloc((size_t)TXT_KSTEPS * NTILES * 32 * 16 * 2);
  (void)ws_size; (void)in_sizes; (void)n_in; (void)out_size;

  // 1) pack weights into WMMA B layout (bf16)
  pack_w_kernel<<<IMG_KSTEPS * NTILES, 32, 0, stream>>>(img_W, IMG_DIM, pImgW);
  pack_w_kernel<<<TXT_KSTEPS * NTILES, 32, 0, stream>>>(txt_W, TXT_DIM, pTxtW);
  // 2) zero pad rows
  zero_pad_kernel<<<1, DIM, 0, stream>>>(imgE, txtE, itemE);
  // 3) fused WMMA projection over the node tables (HBM-bound hot loop)
  table_proj_kernel<<<(MTILES + 7) / 8, 256, 0, stream>>>(
      img_t, txt_t, emb, pImgW, pTxtW, img_b, txt_b, imgE, txtE, itemE);
  // 4) session means
  session_mean_kernel<<<BB, DIM, 0, stream>>>(item, mask, imgE, txtE, itemE,
                                              sImg, sTxt, sItem);
  // 5) gate/qc/softmax fusion
  fusion_kernel<<<BB, DIM, 0, stream>>>(sImg, sTxt, sItem, gvW, gvb, gtW, gtb,
                                        qcW1, qcb1, qcW2, sDiff);
  // 6) gather h0 into d_out
  gather_h_kernel<<<BB * LL, DIM, 0, stream>>>(inputs, itemE, out);
  // 7) 2-layer hypergraph propagation in LDS, in-place on d_out
  propagate_kernel<<<BB, 256, 0, stream>>>(Hs, sDiff, out);
}
